// Model_53463752901248
// MI455X (gfx1250) — matrix operand order repaired and checked by reading, not compiled, not run on hardware
//
#include <hip/hip_runtime.h>
#include <math.h>

// ---------------------------------------------------------------------------
// Problem constants (from reference)
// ---------------------------------------------------------------------------
constexpr int Bc      = 2;
constexpr int Nc      = 20000;
constexpr int NEc     = 120000;
constexpr int ENCc    = 256;
constexpr int HEADSc  = 8;
constexpr int NTOKc   = 256;
constexpr int NBLOCKc = 4;
constexpr int ENCSc   = 33;                 // SPACE + 2*POS_ENC*SPACE
constexpr int ROWS_N  = Bc * Nc;            // 40000 node rows
constexpr int ROWS_E  = Bc * NEc;           // 240000 edge rows

// ---------------------------------------------------------------------------
// WMMA types / helpers (CDNA5 gfx1250, wave32)
// ---------------------------------------------------------------------------
typedef __attribute__((ext_vector_type(16))) __bf16 v16bf;
typedef __attribute__((ext_vector_type(8)))  float  v8f;

union FragAB { v16bf v; unsigned int u[8]; };

// pack two f32 (rounded to bf16, round-half-up) into one u32 via v_perm_b32
static __device__ __forceinline__ unsigned int pack_bf16(float lo, float hi) {
  const unsigned int u0 = __float_as_uint(lo) + 0x8000u;
  const unsigned int u1 = __float_as_uint(hi) + 0x8000u;
  return __builtin_amdgcn_perm(u1, u0, 0x07060302u); // {u1.b3,u1.b2,u0.b3,u0.b2}
}
static __device__ __forceinline__ unsigned short f32_bf16s(float f) {
  return (unsigned short)((__float_as_uint(f) + 0x8000u) >> 16);
}

// ---------------------------------------------------------------------------
// GEMM: C[M,N] = act( concatK(A0/A1/A2)[M,K] @ W[K,N] + bias ) (+ Cres)
//   MULTI=true : A = up to 3 K-segments, each with optional per-row gather
//                (indices cached in LDS). Used for edge/node/concat GEMMs.
//   MULTI=false: A = single contiguous row-major matrix (ld0 == K); staging
//                uses aligned float4 loads when K%4==0 (K=256/512 paths).
//   bf16 inputs (on-the-fly f32->bf16), f32 accumulation in v_wmma.
//   BM=256, BN=128, BK=32, 256 threads = 8 wave32 waves, wave tile = 64x64
//   -> 16 v_wmma_f32_16x16x32_bf16 per wave per K-step.
// ---------------------------------------------------------------------------
#define BMt 256
#define BNt 128
#define BKt 32
#define GEMM_THREADS 256

template <bool SILU, bool MULTI>
__global__ __launch_bounds__(GEMM_THREADS)
void gemm_bf16_kernel(const float* __restrict__ A0, const int* __restrict__ I0, int ld0, int k0end,
                      const float* __restrict__ A1, const int* __restrict__ I1, int ld1, int k1end,
                      const float* __restrict__ A2, const int* __restrict__ I2, int ld2,
                      const float* __restrict__ W,  const float* __restrict__ bias,
                      const float* __restrict__ Cres, float* __restrict__ C,
                      int M, int N, int K)
{
  __shared__ unsigned short Asl[BMt][BKt + 4];   // [m][k], pitch 36 (8B-aligned chunks)
  __shared__ unsigned short Wsl[BNt][BKt + 4];   // transposed [n][k]
  __shared__ int rowIdx[MULTI ? 3 : 1][BMt];     // resolved gather rows per segment

  const int tid  = threadIdx.x;
  const int lane = tid & 31;
  const int wid  = tid >> 5;
  const int half = lane >> 4;
  const int l16  = lane & 15;
  const int wm0  = (wid >> 1) * 64;              // 4 wave-rows
  const int wn0  = (wid & 1) * 64;               // 2 wave-cols
  const int bm   = blockIdx.x * BMt;
  const int bn   = blockIdx.y * BNt;

  // ---- preload per-row gather indices once per block (MULTI only) ----
  if (MULTI) {
    for (int r = tid; r < BMt; r += GEMM_THREADS) {
      const int gr = bm + r;
      const int ok = (gr < M);
      const int gid = ok ? gr : 0;
      rowIdx[0][r] = (I0 && ok) ? I0[gr] : gid;
      rowIdx[1][r] = (I1 && ok) ? I1[gr] : gid;
      rowIdx[2][r] = (I2 && ok) ? I2[gr] : gid;
    }
  }

  v8f acc[4][4] = {};

  const int nkt = (K + BKt - 1) / BKt;
  for (int kt = 0; kt < nkt; ++kt) {
    const int kbase = kt * BKt;
    __syncthreads();

    // ---- stage A tile: 4-wide k chunks, one ds_store_b64 per chunk ----
    // idx -> (row r = idx>>3, chunk c = idx&7); lanes 0..7 cover one row's
    // 32 k values -> 128B contiguous global traffic per row.
    for (int idx = tid; idx < BMt * 8; idx += GEMM_THREADS) {
      const int r  = idx >> 3;
      const int c  = idx & 7;
      const int gr = bm + r;
      const int gk = kbase + c * 4;
      unsigned int p0, p1;
      if (MULTI) {
        const int s0 = (gk < k0end) ? 0 : ((gk < k1end) ? 1 : 2);
        const int s3 = (gk + 3 < k0end) ? 0 : ((gk + 3 < k1end) ? 1 : 2);
        if (gr < M && gk + 3 < K && s0 == s3) {
          const float* src; int ld, kk;
          if (s0 == 0)      { src = A0; ld = ld0; kk = gk; }
          else if (s0 == 1) { src = A1; ld = ld1; kk = gk - k0end; }
          else              { src = A2; ld = ld2; kk = gk - k1end; }
          const float* p = src + (size_t)rowIdx[s0][r] * ld + kk;
          p0 = pack_bf16(p[0], p[1]);
          p1 = pack_bf16(p[2], p[3]);
        } else {
          unsigned short h[4];
#pragma unroll
          for (int j = 0; j < 4; ++j) {
            const int gkj = gk + j;
            float v = 0.f;
            if (gr < M && gkj < K) {
              const float* src; int ld, kk, s;
              if (gkj < k0end)      { src = A0; ld = ld0; kk = gkj; s = 0; }
              else if (gkj < k1end) { src = A1; ld = ld1; kk = gkj - k0end; s = 1; }
              else                  { src = A2; ld = ld2; kk = gkj - k1end; s = 2; }
              v = src[(size_t)rowIdx[s][r] * ld + kk];
            }
            h[j] = f32_bf16s(v);
          }
          p0 = ((unsigned int)h[1] << 16) | h[0];
          p1 = ((unsigned int)h[3] << 16) | h[2];
        }
      } else {
        // single contiguous segment: ld0 == K, identity rows
        if (gr < M && gk + 3 < K) {
          const float* p = A0 + (size_t)gr * ld0 + gk;
          if ((K & 3) == 0) {                 // 16B-aligned fast path
            const float4 f = *(const float4*)p;
            p0 = pack_bf16(f.x, f.y);
            p1 = pack_bf16(f.z, f.w);
          } else {
            p0 = pack_bf16(p[0], p[1]);
            p1 = pack_bf16(p[2], p[3]);
          }
        } else {
          unsigned short h[4];
#pragma unroll
          for (int j = 0; j < 4; ++j) {
            const int gkj = gk + j;
            const float v = (gr < M && gkj < K) ? A0[(size_t)gr * ld0 + gkj] : 0.f;
            h[j] = f32_bf16s(v);
          }
          p0 = ((unsigned int)h[1] << 16) | h[0];
          p1 = ((unsigned int)h[3] << 16) | h[2];
        }
      }
      unsigned int* dst = (unsigned int*)&Asl[r][c * 4];
      dst[0] = p0;
      dst[1] = p1;
    }

    // ---- stage W tile transposed: idx -> (chunk c = idx>>7, n = idx&127) ----
    // lanes read consecutive n at fixed k -> coalesced; write [n][k] LDS.
    for (int idx = tid; idx < BNt * 8; idx += GEMM_THREADS) {
      const int c  = idx >> 7;
      const int n  = idx & 127;
      const int gn = bn + n;
      const int gk = kbase + c * 4;
      unsigned int p0, p1;
      if (gn < N && gk + 3 < K) {
        const float* p = W + (size_t)gk * N + gn;
        p0 = pack_bf16(p[0], p[(size_t)N]);
        p1 = pack_bf16(p[(size_t)2 * N], p[(size_t)3 * N]);
      } else {
        unsigned short h[4];
#pragma unroll
        for (int j = 0; j < 4; ++j) {
          const int gkj = gk + j;
          const float v = (gn < N && gkj < K) ? W[(size_t)gkj * N + gn] : 0.f;
          h[j] = f32_bf16s(v);
        }
        p0 = ((unsigned int)h[1] << 16) | h[0];
        p1 = ((unsigned int)h[3] << 16) | h[2];
      }
      unsigned int* dst = (unsigned int*)&Wsl[n][c * 4];
      dst[0] = p0;
      dst[1] = p1;
    }
    __syncthreads();

    // ---- fragments per ISA 16-bit A (16x32) / B (32x16) layouts ----
    FragAB af[4], bfr[4];
#pragma unroll
    for (int t = 0; t < 4; ++t) {
      const int m = wm0 + t * 16 + l16;
#pragma unroll
      for (int v = 0; v < 8; ++v) {
        const int kk = (v < 4 ? 2 * v : 16 + 2 * (v - 4)) + half * 8;
        af[t].u[v] = *(const unsigned int*)&Asl[m][kk];
      }
      const int n = wn0 + t * 16 + l16;
#pragma unroll
      for (int v = 0; v < 8; ++v) {
        const int kk = (v < 4 ? 2 * v : 16 + 2 * (v - 4)) + half * 8;
        bfr[t].u[v] = *(const unsigned int*)&Wsl[n][kk];
      }
    }
#pragma unroll
    for (int tm = 0; tm < 4; ++tm)
#pragma unroll
      for (int tn = 0; tn < 4; ++tn)
        acc[tm][tn] = __builtin_amdgcn_wmma_f32_16x16x32_bf16(
            false, af[tm].v, false, bfr[tn].v, (short)0, acc[tm][tn], false, false);
  }

  // ---- epilogue: bias, activation, residual, store ----
#pragma unroll
  for (int tm = 0; tm < 4; ++tm) {
#pragma unroll
    for (int tn = 0; tn < 4; ++tn) {
      const int gn = bn + wn0 + tn * 16 + l16;
      if (gn >= N) continue;
      const float bv = bias ? bias[gn] : 0.f;
#pragma unroll
      for (int i = 0; i < 8; ++i) {
        const int gm = bm + wm0 + tm * 16 + half * 8 + i;
        if (gm >= M) continue;
        float r = acc[tm][tn][i] + bv;
        if (SILU) r = r / (1.f + __expf(-r));
        const size_t o = (size_t)gm * N + gn;
        if (Cres) r += Cres[o];
        C[o] = r;
      }
    }
  }
}

// ---------------------------------------------------------------------------
// LayerNorm over 256 features; one wave per row.
//   O (optional): normalized output.   R (optional): R += normalized output.
// ---------------------------------------------------------------------------
__global__ __launch_bounds__(256)
void ln_kernel(const float* __restrict__ X, const float* __restrict__ g,
               const float* __restrict__ b, float* __restrict__ O,
               float* __restrict__ R, int rows)
{
  const int wid  = threadIdx.x >> 5;
  const int lane = threadIdx.x & 31;
  const int row  = blockIdx.x * 8 + wid;
  if (row >= rows) return;
  const float* x = X + (size_t)row * 256;
  float v[8], s = 0.f, s2 = 0.f;
#pragma unroll
  for (int i = 0; i < 8; ++i) {
    v[i] = x[lane + 32 * i];
    s += v[i];
    s2 += v[i] * v[i];
  }
#pragma unroll
  for (int o = 16; o > 0; o >>= 1) {
    s  += __shfl_xor(s, o, 32);
    s2 += __shfl_xor(s2, o, 32);
  }
  const float mean = s * (1.f / 256.f);
  const float var  = s2 * (1.f / 256.f) - mean * mean;
  const float rstd = rsqrtf(var + 1e-5f);
#pragma unroll
  for (int i = 0; i < 8; ++i) {
    const int c = lane + 32 * i;
    const float y = (v[i] - mean) * rstd * g[c] + b[c];
    const size_t o = (size_t)row * 256 + c;
    if (O) O[o] = y;
    if (R) R[o] += y;
  }
}

// ---------------------------------------------------------------------------
// Scatter-add: agg[recvg[e], :] += eFeat[e, :]
// ---------------------------------------------------------------------------
__global__ void scatter_add_kernel(const float* __restrict__ e,
                                   const int* __restrict__ recvg,
                                   float* __restrict__ agg)
{
  const long t = (long)blockIdx.x * 256 + threadIdx.x;
  if (t >= (long)ROWS_E * 256) return;
  const long ei = t >> 8;
  const int  c  = (int)(t & 255);
  atomicAdd(&agg[(size_t)recvg[ei] * 256 + c], e[t]);
}

// ---------------------------------------------------------------------------
// Edge preprocessing: global row indices + [d1, -d1, |d1|] features
// ---------------------------------------------------------------------------
__global__ void build_edges_kernel(const int* __restrict__ edges,
                                   const float* __restrict__ pos,
                                   int* __restrict__ sendg, int* __restrict__ recvg,
                                   float* __restrict__ feat7)
{
  const int t = blockIdx.x * 256 + threadIdx.x;
  if (t >= ROWS_E) return;
  const int b  = t / NEc;
  const int s  = edges[2 * t];
  const int r  = edges[2 * t + 1];
  const int gs = b * Nc + s;
  const int gr = b * Nc + r;
  sendg[t] = gs;
  recvg[t] = gr;
  const float dx = pos[gr * 3 + 0] - pos[gs * 3 + 0];
  const float dy = pos[gr * 3 + 1] - pos[gs * 3 + 1];
  const float dz = pos[gr * 3 + 2] - pos[gs * 3 + 2];
  float* f = feat7 + (size_t)t * 7;
  f[0] = dx; f[1] = dy; f[2] = dz;
  f[3] = -dx; f[4] = -dy; f[5] = -dz;
  f[6] = sqrtf(dx * dx + dy * dy + dz * dz);
}

// ---------------------------------------------------------------------------
// Fourier positional encoding: (B*N, 3) -> (B*N, 33) [x, sin, cos]
// ---------------------------------------------------------------------------
__global__ void fourier_kernel(const float* __restrict__ pos, float* __restrict__ senc)
{
  const int t = blockIdx.x * 256 + threadIdx.x;
  if (t >= ROWS_N) return;
  const float PI = 3.14159265358979323846f;
  float p[3] = { pos[t * 3], pos[t * 3 + 1], pos[t * 3 + 2] };
  float* o = senc + (size_t)t * ENCSc;
  o[0] = p[0]; o[1] = p[1]; o[2] = p[2];
#pragma unroll
  for (int s = 0; s < 3; ++s)
#pragma unroll
    for (int k = 0; k < 5; ++k) {
      const float ang = p[s] * (float)(1 << k) * PI;
      o[3 + s * 5 + k]  = sinf(ang);
      o[18 + s * 5 + k] = cosf(ang);
    }
}

// ---------------------------------------------------------------------------
// Vin = [V | s_enc]   (B*N, 289)
// ---------------------------------------------------------------------------
__global__ void concat_vin_kernel(const float* __restrict__ V,
                                  const float* __restrict__ senc,
                                  float* __restrict__ Vin)
{
  const long t = (long)blockIdx.x * 256 + threadIdx.x;
  if (t >= (long)ROWS_N * 289) return;
  const long r = t / 289;
  const int  c = (int)(t % 289);
  Vin[t] = (c < 256) ? V[r * 256 + c] : senc[r * 33 + (c - 256)];
}

// ---------------------------------------------------------------------------
// Attention stage 1: 256 latents attend over N nodes (per b,h).
//   lane == head-dim (d=32 == wave32). Online softmax, K/V tiles in LDS
//   shared by 16 query-waves per workgroup.
// ---------------------------------------------------------------------------
__global__ __launch_bounds__(512)
void attn1_kernel(const float* __restrict__ Q1, const float* __restrict__ Kx,
                  const float* __restrict__ Vx, float* __restrict__ Z)
{
  __shared__ float Ks[128][32];
  __shared__ float Vs[128][32];
  const int bid  = blockIdx.x;           // b*128 + h*16 + qc
  const int qc   = bid & 15;
  const int h    = (bid >> 4) & 7;
  const int b    = bid >> 7;
  const int tid  = threadIdx.x;
  const int lane = tid & 31;
  const int w    = tid >> 5;
  const int q    = qc * 16 + w;
  const float qv = Q1[(size_t)q * 256 + h * 32 + lane];  // latents: no batch dim
  const float scale = 0.1767766952966369f;               // 1/sqrt(32)
  float m = -1e30f, l = 0.f, acc = 0.f;

  for (int nt = 0; nt < Nc; nt += 128) {
    const int count = min(128, Nc - nt);
    __syncthreads();
    for (int idx = tid; idx < count * 32; idx += 512) {
      const int j = idx >> 5, d = idx & 31;
      const size_t off = ((size_t)(b * Nc + nt + j)) * 256 + h * 32 + d;
      Ks[j][d] = Kx[off];
      Vs[j][d] = Vx[off];
    }
    __syncthreads();
    for (int j = 0; j < count; ++j) {
      float p = qv * Ks[j][lane];
#pragma unroll
      for (int o = 16; o > 0; o >>= 1) p += __shfl_xor(p, o, 32);
      const float s  = p * scale;
      const float mn = fmaxf(m, s);
      const float cr = __expf(m - mn);
      const float e  = __expf(s - mn);
      l   = l * cr + e;
      acc = acc * cr + e * Vs[j][lane];
      m   = mn;
    }
  }
  Z[((size_t)(b * NTOKc + q)) * 256 + h * 32 + lane] = acc / l;
}

// ---------------------------------------------------------------------------
// Attention stage 2: N nodes attend over 256 latents (per b,h).
//   All 256 latent K/V rows for (b,h) staged once in LDS (64 KB).
// ---------------------------------------------------------------------------
#define A2_QBLOCKS 40
__global__ __launch_bounds__(256)
void attn2_kernel(const float* __restrict__ Q2, const float* __restrict__ Kz,
                  const float* __restrict__ Vz, float* __restrict__ Out)
{
  __shared__ float Ks[256][32];
  __shared__ float Vs[256][32];
  const int bid  = blockIdx.x;           // b*(8*A2_QBLOCKS) + h*A2_QBLOCKS + qc
  const int qc   = bid % A2_QBLOCKS;
  const int h    = (bid / A2_QBLOCKS) & 7;
  const int b    = bid / (A2_QBLOCKS * 8);
  const int tid  = threadIdx.x;
  const int lane = tid & 31;
  const int w    = tid >> 5;

  for (int idx = tid; idx < 256 * 32; idx += 256) {
    const int j = idx >> 5, d = idx & 31;
    const size_t off = ((size_t)(b * NTOKc + j)) * 256 + h * 32 + d;
    Ks[j][d] = Kz[off];
    Vs[j][d] = Vz[off];
  }
  __syncthreads();

  const int per = (Nc + A2_QBLOCKS - 1) / A2_QBLOCKS;
  const int q0 = qc * per;
  const int q1 = min(Nc, q0 + per);
  const float scale = 0.1767766952966369f;

  for (int q = q0 + w; q < q1; q += 8) {
    const float qv = Q2[((size_t)(b * Nc + q)) * 256 + h * 32 + lane];
    float m = -1e30f, l = 0.f, acc = 0.f;
    for (int j = 0; j < 256; ++j) {
      float p = qv * Ks[j][lane];
#pragma unroll
      for (int o = 16; o > 0; o >>= 1) p += __shfl_xor(p, o, 32);
      const float s  = p * scale;
      const float mn = fmaxf(m, s);
      const float cr = __expf(m - mn);
      const float e  = __expf(s - mn);
      l   = l * cr + e;
      acc = acc * cr + e * Vs[j][lane];
      m   = mn;
    }
    Out[((size_t)(b * Nc + q)) * 256 + h * 32 + lane] = acc / l;
  }
}

// ---------------------------------------------------------------------------
// Host-side orchestration
// ---------------------------------------------------------------------------
// Input flattening (setup_inputs() dict insertion order, nested dicts in
// insertion order, _lin = {W,b}, _mlp = {l1.W,l1.b,l2.W,l2.b[,g,bt]}):
//   0 state_in  1 node_pos  2 edges
//   3..6 enc, 7..10 fe, 11..14 dec_delta, 15..18 dec_state
//   19 + 28*blk + { 0..5: ge, 6..11: gn, 12..15: ln1g,ln1b,ln2g,ln2b,
//                   16: lat, 17..23: Wq1,Wk1,Wv1,Wq2,Wk2,Wv2,Wo,
//                   24..27: ffn1.W, ffn1.b, ffn2.W, ffn2.b }

extern "C" void kernel_launch(void* const* d_in, const int* in_sizes, int n_in,
                              void* d_out, int out_size, void* d_ws, size_t ws_size,
                              hipStream_t stream)
{
  (void)in_sizes; (void)out_size; (void)ws_size;

  const float* state_in = (const float*)d_in[0];
  const float* node_pos = (const float*)d_in[1];
  const int*   edges    = (const int*)d_in[2];
  const float* P[131];
  for (int i = 3; i < n_in && i < 131; ++i) P[i] = (const float*)d_in[i];

  // ---- workspace bump allocator ----
  char* wptr = (char*)d_ws;
  auto alloc = [&](size_t bytes) -> void* {
    void* p = (void*)wptr;
    wptr += (bytes + 255) & ~(size_t)255;
    return p;
  };
  float* senc  = (float*)alloc((size_t)ROWS_N * ENCSc * 4);
  float* V     = (float*)alloc((size_t)ROWS_N * 256 * 4);
  float* Vin   = (float*)alloc((size_t)ROWS_N * 289 * 4);
  float* E     = (float*)alloc((size_t)ROWS_E * 256 * 4);
  float* W1    = (float*)alloc((size_t)ROWS_E * 256 * 4);   // edge hidden
  float* W2    = (float*)alloc((size_t)ROWS_E * 256 * 4);   // edge out / e
  float* agg   = (float*)alloc((size_t)ROWS_N * 256 * 4);
  float* T1    = (float*)alloc((size_t)ROWS_N * 512 * 4);   // node hidden (ffn wide)
  float* T2    = (float*)alloc((size_t)ROWS_N * 256 * 4);   // pre-LN / xln / h
  float* T3    = (float*)alloc((size_t)ROWS_N * 256 * 4);   // Kx
  float* T4    = (float*)alloc((size_t)ROWS_N * 256 * 4);   // Vx
  float* T5    = (float*)alloc((size_t)ROWS_N * 256 * 4);   // Q2x
  float* T6    = (float*)alloc((size_t)ROWS_N * 256 * 4);   // attn out
  float* feat7 = (float*)alloc((size_t)ROWS_E * 7 * 4);
  int*   sendg = (int*)alloc((size_t)ROWS_E * 4);
  int*   recvg = (int*)alloc((size_t)ROWS_E * 4);
  float* zq1   = (float*)alloc((size_t)NTOKc * 256 * 4);
  float* zbuf  = (float*)alloc((size_t)Bc * NTOKc * 256 * 4);
  float* zk    = (float*)alloc((size_t)Bc * NTOKc * 256 * 4);
  float* zv    = (float*)alloc((size_t)Bc * NTOKc * 256 * 4);

  // ---- GEMM launch helpers ----
  auto gemm3 = [&](const float* A0, const int* I0, int ld0, int k0,
                   const float* A1, const int* I1, int ld1, int k1,
                   const float* A2, const int* I2, int ld2,
                   const float* W, const float* bias, const float* Cres, float* C,
                   int M, int N, int K, bool silu) {
    dim3 grid((M + BMt - 1) / BMt, (N + BNt - 1) / BNt);
    if (silu)
      gemm_bf16_kernel<true, true><<<grid, GEMM_THREADS, 0, stream>>>(
          A0, I0, ld0, k0, A1, I1, ld1, k1, A2, I2, ld2, W, bias, Cres, C, M, N, K);
    else
      gemm_bf16_kernel<false, true><<<grid, GEMM_THREADS, 0, stream>>>(
          A0, I0, ld0, k0, A1, I1, ld1, k1, A2, I2, ld2, W, bias, Cres, C, M, N, K);
  };
  auto gemm1 = [&](const float* A, const float* W, const float* bias,
                   const float* Cres, float* C, int M, int N, int K, bool silu) {
    dim3 grid((M + BMt - 1) / BMt, (N + BNt - 1) / BNt);
    if (silu)
      gemm_bf16_kernel<true, false><<<grid, GEMM_THREADS, 0, stream>>>(
          A, nullptr, K, K, nullptr, nullptr, 0, K, nullptr, nullptr, 0,
          W, bias, Cres, C, M, N, K);
    else
      gemm_bf16_kernel<false, false><<<grid, GEMM_THREADS, 0, stream>>>(
          A, nullptr, K, K, nullptr, nullptr, 0, K, nullptr, nullptr, 0,
          W, bias, Cres, C, M, N, K);
  };
  auto ln = [&](const float* X, const float* g, const float* b,
                float* O, float* R, int rows) {
    ln_kernel<<<(rows + 7) / 8, 256, 0, stream>>>(X, g, b, O, R, rows);
  };

  // ---- preprocessing ----
  build_edges_kernel<<<(ROWS_E + 255) / 256, 256, 0, stream>>>(edges, node_pos, sendg, recvg, feat7);
  fourier_kernel<<<(ROWS_N + 255) / 256, 256, 0, stream>>>(node_pos, senc);

  // encoder: V = l2( silu( [state|pos] @ l1 ) )
  gemm3(state_in, nullptr, 4, 4, node_pos, nullptr, 3, 7, nullptr, nullptr, 0,
        P[3], P[4], nullptr, T2, ROWS_N, 256, 7, true);
  gemm1(T2, P[5], P[6], nullptr, V, ROWS_N, 256, 256, false);

  // edge encoder fe: E = l2( silu( feat7 @ l1 ) )
  gemm1(feat7, P[7], P[8], nullptr, W1, ROWS_E, 256, 7, true);
  gemm1(W1, P[9], P[10], nullptr, E, ROWS_E, 256, 256, false);

  // ---- message-passing blocks ----
  for (int blk = 0; blk < NBLOCKc; ++blk) {
    const int pb = 19 + 28 * blk;

    concat_vin_kernel<<<(int)(((long)ROWS_N * 289 + 255) / 256), 256, 0, stream>>>(V, senc, Vin);

    // edge MLP: h = silu([Vin[send]|Vin[recv]|E] @ ge.l1); e_pre = h @ ge.l2
    gemm3(Vin, sendg, 289, 289, Vin, recvg, 289, 578, E, nullptr, 256,
          P[pb + 0], P[pb + 1], nullptr, W1, ROWS_E, 256, 834, true);
    gemm1(W1, P[pb + 2], P[pb + 3], nullptr, W2, ROWS_E, 256, 256, false);
    // e = LN(e_pre) ; E += e
    ln(W2, P[pb + 4], P[pb + 5], W2, E, ROWS_E);

    // agg = scatter_add(e, recv)
    hipMemsetAsync(agg, 0, (size_t)ROWS_N * 256 * 4, stream);
    scatter_add_kernel<<<(int)(((long)ROWS_E * 256 + 255) / 256), 256, 0, stream>>>(W2, recvg, agg);

    // node MLP: h = silu([Vin|agg] @ gn.l1); v_pre = h @ gn.l2; V += LN(v_pre)
    gemm3(Vin, nullptr, 289, 289, agg, nullptr, 256, 545, nullptr, nullptr, 0,
          P[pb + 6], P[pb + 7], nullptr, T1, ROWS_N, 256, 545, true);
    gemm1(T1, P[pb + 8], P[pb + 9], nullptr, T2, ROWS_N, 256, 256, false);
    ln(T2, P[pb + 10], P[pb + 11], nullptr, V, ROWS_N);

    // attention: x = LN(V)
    ln(V, P[pb + 12], P[pb + 13], T2, nullptr, ROWS_N);
    gemm1(P[pb + 16], P[pb + 17], nullptr, nullptr, zq1, NTOKc, 256, 256, false); // lat@Wq1
    gemm1(T2, P[pb + 18], nullptr, nullptr, T3, ROWS_N, 256, 256, false);          // x@Wk1
    gemm1(T2, P[pb + 19], nullptr, nullptr, T4, ROWS_N, 256, 256, false);          // x@Wv1
    gemm1(T2, P[pb + 20], nullptr, nullptr, T5, ROWS_N, 256, 256, false);          // x@Wq2
    attn1_kernel<<<Bc * HEADSc * 16, 512, 0, stream>>>(zq1, T3, T4, zbuf);
    gemm1(zbuf, P[pb + 21], nullptr, nullptr, zk, Bc * NTOKc, 256, 256, false);    // z@Wk2
    gemm1(zbuf, P[pb + 22], nullptr, nullptr, zv, Bc * NTOKc, 256, 256, false);    // z@Wv2
    attn2_kernel<<<Bc * HEADSc * A2_QBLOCKS, 256, 0, stream>>>(T5, zk, zv, T6);
    gemm1(T6, P[pb + 23], nullptr, V, V, ROWS_N, 256, 256, false);                 // V += attn@Wo

    // FFN: h = LN(V); V += silu(h@ffn1) @ ffn2
    ln(V, P[pb + 14], P[pb + 15], T2, nullptr, ROWS_N);
    gemm1(T2, P[pb + 24], P[pb + 25], nullptr, T1, ROWS_N, 512, 256, true);
    gemm1(T1, P[pb + 26], P[pb + 27], V, V, ROWS_N, 256, 512, false);
  }

  // decoder delta: V += l2( silu( [V|s_enc] @ l1 ) )
  gemm3(V, nullptr, 256, 256, senc, nullptr, 33, 289, nullptr, nullptr, 0,
        P[11], P[12], nullptr, T1, ROWS_N, 256, 289, true);
  gemm1(T1, P[13], P[14], V, V, ROWS_N, 256, 256, false);

  // decoder state: out = l2( silu( V @ l1 ) ), N=3
  gemm1(V, P[15], P[16], nullptr, T1, ROWS_N, 256, 256, true);
  gemm1(T1, P[17], P[18], nullptr, (float*)d_out, ROWS_N, 3, 256, false);
}
